// RNN_34479997453077
// MI455X (gfx1250) — compile-verified, matrix operand order checked by reading, not run on hardware
//
#include <hip/hip_runtime.h>
#include <cmath>

// Problem constants (from reference): T=512, B=64, IN=512, LAT=1024
#define T_STEPS   512
#define BATCH     64
#define IN_DIM    512
#define LAT       1024
#define NWG       64               // scan: one WG per 16-column slice of LAT
#define NTHREADS  128              // 4 wave32s, one per 16-row tile of BATCH
#define WH_STRIDE 1032             // 1024 + 8 pad -> conflict-free ds_load_b128
#define WX_STRIDE 520              // 512 + 8 pad

typedef __attribute__((ext_vector_type(16))) __bf16 v16bf;
typedef __attribute__((ext_vector_type(8)))  float  v8f;

__device__ __forceinline__ unsigned short f2bf(float f) {
  __bf16 h = (__bf16)f;                       // v_cvt_pk_bf16_f32 when paired
  return __builtin_bit_cast(unsigned short, h);
}

__device__ __forceinline__ float fast_tanh(float f) {
#if __has_builtin(__builtin_amdgcn_tanhf)
  return __builtin_amdgcn_tanhf(f);           // hardware V_TANH_F32
#else
  return tanhf(f);
#endif
}

union FragU {
  v16bf v;
  uint4 q[2];
  unsigned short s[16];
};

// A/B fragment for v_wmma_f32_16x16x32_bf16, per documented gfx1250 layout:
// lane half (lane>>4) selects K offset +8; VGPRs 0-3 hold K=k0+h*8..+7,
// VGPRs 4-7 hold K=k0+16+h*8..+7. Two b128 loads per lane per 32-K chunk.
__device__ __forceinline__ v16bf frag_from_bf16(const unsigned short* rowbase,
                                                int k0, int half) {
  FragU u;
  u.q[0] = *(const uint4*)(rowbase + k0 + half * 8);
  u.q[1] = *(const uint4*)(rowbase + k0 + 16 + half * 8);
  return u.v;
}

__device__ __forceinline__ v16bf frag_from_f32(const float* rowbase,
                                               int k0, int half) {
  const float4* p0 = (const float4*)(rowbase + k0 + half * 8);
  const float4* p1 = (const float4*)(rowbase + k0 + 16 + half * 8);
  float4 a0 = p0[0], a1 = p0[1];
  float4 b0 = p1[0], b1 = p1[1];
  FragU u;
  u.s[0]  = f2bf(a0.x); u.s[1]  = f2bf(a0.y);
  u.s[2]  = f2bf(a0.z); u.s[3]  = f2bf(a0.w);
  u.s[4]  = f2bf(a1.x); u.s[5]  = f2bf(a1.y);
  u.s[6]  = f2bf(a1.z); u.s[7]  = f2bf(a1.w);
  u.s[8]  = f2bf(b0.x); u.s[9]  = f2bf(b0.y);
  u.s[10] = f2bf(b0.z); u.s[11] = f2bf(b0.w);
  u.s[12] = f2bf(b1.x); u.s[13] = f2bf(b1.y);
  u.s[14] = f2bf(b1.z); u.s[15] = f2bf(b1.w);
  return u.v;
}

// -------------------------------------------------------------------------
// Kernel 0: h0 -> bf16 ping buffer, zero per-step barrier counters (re-run
// every call so graph replay stays deterministic).
// -------------------------------------------------------------------------
__global__ void rnn_init_kernel(const float* __restrict__ h0,
                                unsigned short* __restrict__ hbuf,
                                unsigned* __restrict__ bar) {
  int i = blockIdx.x * blockDim.x + threadIdx.x;
  if (i < BATCH * LAT) hbuf[i] = f2bf(h0[i]);
  if (i < T_STEPS)     bar[i]  = 0u;
}

// -------------------------------------------------------------------------
// Kernel 1: xw = x @ Wx + b, written INTO d_out (same shape: T*B x LAT f32).
// Fully parallel: grid = (LAT/16 col blocks, T*B/64 row blocks), 4 waves/WG,
// each wave one 16x16 tile over K=512 (16 bf16 WMMA, 2 interleaved acc
// chains). Wx slice staged in LDS once per WG.
// -------------------------------------------------------------------------
__global__ __launch_bounds__(NTHREADS)
void xw_gemm_kernel(const float* __restrict__ x,
                    const float* __restrict__ W,
                    const float* __restrict__ b,
                    float* __restrict__ out) {
  __shared__ __align__(16) unsigned short wlds[16 * WX_STRIDE];

  const int col0 = blockIdx.x * 16;
  const int row0 = blockIdx.y * 64;      // flattened (t*B + batch) row block
  const int tid  = threadIdx.x;
  const int wrow = tid >> 5;
  const int lane = tid & 31;
  const int n    = lane & 15;
  const int half = lane >> 4;
  const int mrow = row0 + wrow * 16 + n;

  // Stage Wx[:,col0:col0+16] transposed into LDS (f32 -> bf16).
  for (int idx = tid; idx < 16 * IN_DIM; idx += NTHREADS) {
    int nn = idx & 15;
    int k  = idx >> 4;
    wlds[nn * WX_STRIDE + k] = f2bf(W[(size_t)k * LAT + col0 + nn]);
  }
  __syncthreads();

  const float bval = b[col0 + n];
  const unsigned short* wcol = &wlds[n * WX_STRIDE];
  const float* xrow = x + (size_t)mrow * IN_DIM;

  v8f acc0, acc1;
  #pragma unroll
  for (int j = 0; j < 8; ++j) { acc0[j] = bval; acc1[j] = 0.0f; }

  #pragma unroll 4
  for (int c = 0; c < IN_DIM / 32; c += 2) {
    v16bf a0 = frag_from_f32(xrow, c * 32, half);
    v16bf b0 = frag_from_bf16(wcol, c * 32, half);
    acc0 = __builtin_amdgcn_wmma_f32_16x16x32_bf16(
             false, a0, false, b0, (short)0, acc0, false, false);
    v16bf a1 = frag_from_f32(xrow, (c + 1) * 32, half);
    v16bf b1 = frag_from_bf16(wcol, (c + 1) * 32, half);
    acc1 = __builtin_amdgcn_wmma_f32_16x16x32_bf16(
             false, a1, false, b1, (short)0, acc1, false, false);
  }
  v8f acc = acc0 + acc1;

  // C/D layout: VGPR j -> row = 8*half + j (within tile), col = n.
  #pragma unroll
  for (int j = 0; j < 8; ++j) {
    int row = row0 + wrow * 16 + half * 8 + j;
    out[(size_t)row * LAT + col0 + n] = acc[j];
  }
}

// -------------------------------------------------------------------------
// Kernel 2: sequential scan h_t = tanh(xw_t + h_{t-1} @ Wh), overwriting
// out[t] (which holds xw_t) with h_t in place. 64 persistent WGs; Wh slice
// (1024x16 bf16, 33 KB) lives in LDS for all 512 steps; h ping-pongs through
// L2 in bf16; device-wide atomic barrier per step. Four independent WMMA
// accumulator chains break the D->C dependency (8 deep instead of 32).
// -------------------------------------------------------------------------
__global__ __launch_bounds__(NTHREADS, 1)
void rnn_scan_kernel(const float* __restrict__ W,
                     float* __restrict__ out,
                     unsigned short* __restrict__ hbuf,
                     unsigned* __restrict__ bar) {
  __shared__ __align__(16) unsigned short wlds[16 * WH_STRIDE];

  const int col0 = blockIdx.x * 16;
  const int tid  = threadIdx.x;
  const int wrow = tid >> 5;
  const int lane = tid & 31;
  const int n    = lane & 15;
  const int half = lane >> 4;
  const int mrow = wrow * 16 + n;        // A-matrix row (batch index 0..63)

  // Stage Wh[:,col0:col0+16] (= W rows 512..1535) transposed into LDS.
  for (int idx = tid; idx < 16 * LAT; idx += NTHREADS) {
    int nn = idx & 15;
    int k  = idx >> 4;
    wlds[nn * WH_STRIDE + k] = f2bf(W[(size_t)(IN_DIM + k) * LAT + col0 + nn]);
  }
  __syncthreads();

  const unsigned short* wcol = &wlds[n * WH_STRIDE];

  for (int t = 0; t < T_STEPS; ++t) {
    const unsigned short* hcur = hbuf + (size_t)(t & 1) * (BATCH * LAT);
    unsigned short*       hnxt = hbuf + (size_t)((t + 1) & 1) * (BATCH * LAT);
    float* outt = out + (size_t)t * BATCH * LAT;

    // Chain 0 starts from xw_t (precomputed into d_out by kernel 1); the
    // loads issue up front and overlap the WMMA chains.
    const float* xwp = outt + (size_t)(wrow * 16 + half * 8) * LAT + col0 + n;
    v8f acc[4];
    #pragma unroll
    for (int j = 0; j < 8; ++j) {
      acc[0][j] = xwp[(size_t)j * LAT];
      acc[1][j] = 0.0f; acc[2][j] = 0.0f; acc[3][j] = 0.0f;
    }

    // h_{t-1} @ Wh : K = 1024 = 32 chunks, round-robined over 4 acc chains.
    const unsigned short* hrowp = hcur + (size_t)mrow * LAT;
    #pragma unroll 4
    for (int c = 0; c < LAT / 32; ++c) {
      v16bf afrag = frag_from_bf16(hrowp, c * 32, half);
      v16bf bfrag = frag_from_bf16(wcol, c * 32, half);
      acc[c & 3] = __builtin_amdgcn_wmma_f32_16x16x32_bf16(
              false, afrag, false, bfrag, (short)0, acc[c & 3], false, false);
    }
    v8f accs = (acc[0] + acc[1]) + (acc[2] + acc[3]);

    // tanh in f32; overwrite xw_t with h_t (f32) and emit next-step h (bf16).
    #pragma unroll
    for (int j = 0; j < 8; ++j) {
      int row = wrow * 16 + half * 8 + j;
      float v = fast_tanh(accs[j]);
      outt[(size_t)row * LAT + col0 + n] = v;
      hnxt[(size_t)row * LAT + col0 + n] = f2bf(v);
    }

    // Device-wide step barrier: make h visible, arrive, spin.
    __threadfence();
    __syncthreads();
    if (tid == 0) {
      __hip_atomic_fetch_add(&bar[t], 1u, __ATOMIC_ACQ_REL,
                             __HIP_MEMORY_SCOPE_AGENT);
      while (__hip_atomic_load(&bar[t], __ATOMIC_ACQUIRE,
                               __HIP_MEMORY_SCOPE_AGENT) < NWG) {
        __builtin_amdgcn_s_sleep(1);
      }
    }
    __syncthreads();
  }
}

extern "C" void kernel_launch(void* const* d_in, const int* in_sizes, int n_in,
                              void* d_out, int out_size, void* d_ws, size_t ws_size,
                              hipStream_t stream) {
  const float* x  = (const float*)d_in[0];  // (T,B,IN) f32
  const float* h0 = (const float*)d_in[1];  // (B,LAT)  f32
  const float* W  = (const float*)d_in[2];  // (IN+LAT,LAT) f32
  const float* b  = (const float*)d_in[3];  // (LAT,)   f32
  float* out = (float*)d_out;               // (T,B,LAT) f32

  // Workspace: 2 bf16 h buffers (ping-pong) + per-step barrier counters.
  unsigned short* hbuf = (unsigned short*)d_ws;
  unsigned* bar = (unsigned*)((char*)d_ws +
                              (size_t)2 * BATCH * LAT * sizeof(unsigned short));

  rnn_init_kernel<<<(BATCH * LAT + 255) / 256, 256, 0, stream>>>(h0, hbuf, bar);

  // Parallel pre-GEMM: xw = x@Wx + b  ->  d_out  (consumed in place by scan).
  dim3 g1(LAT / 16, (T_STEPS * BATCH) / 64, 1);
  xw_gemm_kernel<<<g1, NTHREADS, 0, stream>>>(x, W, b, out);

  // Latency-bound scan over T steps.
  rnn_scan_kernel<<<NWG, NTHREADS, 0, stream>>>(W, out, hbuf, bar);

  (void)in_sizes; (void)n_in; (void)out_size; (void)ws_size;
}
